// PointTransformerSeg_57664230916167
// MI455X (gfx1250) — matrix operand order repaired and checked by reading, not compiled, not run on hardware
//
#include <hip/hip_runtime.h>

// ---------------------------------------------------------------------------
// Problem constants (from reference): B, N, D, L, H, K, NC = 2,8192,96,4,192,16,2
// ---------------------------------------------------------------------------
#define BB 2
#define NN 8192
#define DD 96
#define LL 4
#define HH 192
#define KK 16
#define NCC 2
#define NP 4          // points per attention block / n-tiles per gemm wave
#define EPS 1e-5f

typedef __attribute__((ext_vector_type(16))) __bf16 bfx16;
typedef __attribute__((ext_vector_type(8)))  __bf16 bfx8;
typedef __attribute__((ext_vector_type(8)))  float  fx8;

// ---------------------------------------------------------------------------
// WMMA 16x16 tile GEMM with 4-way N register blocking (A fragment reused):
//   C_t(16x16,f32) += A(16xKd, bf16 row-major) * B_t(Kd x 16)
// B stored column-major rows of length ldb: Bp[col*ldb + k].
// v_wmma_f32_16x16x32_bf16 operand layouts used here (A/B k order chosen by this code, shared by both):
//   A 16x32: lane l -> M = l&15; kh = l>>4; elems e -> K = kt+16*kh+e
//            (contiguous 16, loaded as two 8-elem runs)
//   B 32x16: lane l -> N = l&15; elems e -> K = kt + 16*kh + e (contiguous 16)
//   C/D:     lane l -> N = l&15; vgpr r -> M = r + 8*kh
// ---------------------------------------------------------------------------
__device__ __forceinline__ void wmma_mm4(const __bf16* __restrict__ A, int lda,
                                         const __bf16* __restrict__ Bp, int ldb,
                                         int Kd, int mtile, int ncolbase, int lane,
                                         fx8* acc) {
  const int mrow = mtile * 16 + (lane & 15);
  const int kh   = lane >> 4;
  const int cl   = lane & 15;
  for (int kt = 0; kt < Kd; kt += 32) {
    const __bf16* ap = A + (size_t)mrow * lda + kt + kh * 16;
    bfx8 a0 = *(const bfx8*)(ap);
    bfx8 a1 = *(const bfx8*)(ap + 8);
    bfx16 av;
#pragma unroll
    for (int i = 0; i < 8; ++i) { av[i] = a0[i]; av[i + 8] = a1[i]; }
#pragma unroll
    for (int t = 0; t < NP; ++t) {
      bfx16 bv = *(const bfx16*)(Bp + (size_t)(ncolbase + t * 16 + cl) * ldb +
                                 kt + kh * 16);
      acc[t] = __builtin_amdgcn_wmma_f32_16x16x32_bf16(false, av, false, bv,
                                                       (short)0, acc[t],
                                                       false, false);
    }
  }
}

// ---------------------------------------------------------------------------
// fp32 -> bf16 conversion (weights)
// ---------------------------------------------------------------------------
__global__ void cvt_bf16_kernel(const float* __restrict__ src,
                                __bf16* __restrict__ dst, int n) {
  int i = blockIdx.x * 256 + threadIdx.x;
  if (i < n) dst[i] = (__bf16)src[i];
}

// ---------------------------------------------------------------------------
// Embed: x[b,o,n] = relu(bn(embed_w @ points)); writes fp32 (B,D,N) + bf16 (B,N,D)
// ---------------------------------------------------------------------------
__global__ void embed_kernel(const float* __restrict__ points,
                             const float* __restrict__ ew,
                             const float* __restrict__ eg, const float* __restrict__ eb,
                             const float* __restrict__ em, const float* __restrict__ ev,
                             float* __restrict__ xf, __bf16* __restrict__ xbf) {
  int n = blockIdx.x * blockDim.x + threadIdx.x;
  int o = blockIdx.y, b = blockIdx.z;
  const float* p = points + ((size_t)b * NN + n) * 3;
  float a = ew[o * 3 + 0] * p[0] + ew[o * 3 + 1] * p[1] + ew[o * 3 + 2] * p[2];
  float s = eg[o] * rsqrtf(ev[o] + EPS);
  float y = fmaxf((a - em[o]) * s + eb[o], 0.0f);
  xf[((size_t)b * DD + o) * NN + n] = y;
  xbf[((size_t)b * NN + n) * DD + o] = (__bf16)y;
}

// ---------------------------------------------------------------------------
// Brute-force KNN (top-16 by dist = |q|^2 + |m|^2 - 2 q.m, ties -> lower index)
// ---------------------------------------------------------------------------
__global__ void knn_kernel(const float* __restrict__ points, int* __restrict__ idxw) {
  __shared__ float sp[256 * 3];
  int tid = threadIdx.x;
  int q = blockIdx.x * 256 + tid;            // linear over b*N+n
  int b = q / NN;
  const float* qp = points + (size_t)q * 3;
  float qx = qp[0], qy = qp[1], qz = qp[2];
  float qsq = qx * qx + qy * qy + qz * qz;
  float bd[KK];
  int   bi[KK];
#pragma unroll
  for (int k = 0; k < KK; ++k) { bd[k] = 3.4e38f; bi[k] = 0; }
  for (int m0 = 0; m0 < NN; m0 += 256) {
    __syncthreads();
    const float* mp = points + ((size_t)b * NN + m0 + tid) * 3;
    sp[tid * 3 + 0] = mp[0];
    sp[tid * 3 + 1] = mp[1];
    sp[tid * 3 + 2] = mp[2];
    __syncthreads();
    for (int j = 0; j < 256; ++j) {
      float px = sp[j * 3 + 0], py = sp[j * 3 + 1], pz = sp[j * 3 + 2];
      float msq = px * px + py * py + pz * pz;
      float dist = qsq + msq - 2.0f * (qx * px + qy * py + qz * pz);
      if (dist < bd[KK - 1]) {
        int pos = KK - 1;
        while (pos > 0 && dist < bd[pos - 1]) {
          bd[pos] = bd[pos - 1]; bi[pos] = bi[pos - 1]; --pos;
        }
        bd[pos] = dist; bi[pos] = m0 + j;
      }
    }
  }
#pragma unroll
  for (int k = 0; k < KK; ++k) idxw[(size_t)q * KK + k] = bi[k];
}

// ---------------------------------------------------------------------------
// Generic WMMA GEMM over N=8192 columns: C = A(MxKd) * X(Kd x N per batch).
// Each wave: 1 M-tile x 4 N-tiles (A fragment reused across 4 WMMAs).
//   mode 0: store fp32 C to outF (B,M,N)
//   mode 1: relu(bn(C)) -> bf16 outBF (B,N,M)
//   mode 2: y=bn(resid+C) -> fp32 outF (B,M,N) and bf16 outBF (B,N,M)
// block = 128 threads (4 waves) -> 256 columns per block.
// ---------------------------------------------------------------------------
__global__ void gemm_kernel(const __bf16* __restrict__ A,
                            const __bf16* __restrict__ Xbf,
                            int M, int Kd, int mode,
                            const float* __restrict__ bg, const float* __restrict__ bbp,
                            const float* __restrict__ bm, const float* __restrict__ bvv,
                            const float* __restrict__ resid,
                            float* __restrict__ outF, __bf16* __restrict__ outBF) {
  int lane   = threadIdx.x & 31;
  int wid    = threadIdx.x >> 5;
  int ntile0 = (blockIdx.x * 4 + wid) * NP;   // first of 4 n-tiles for this wave
  int mtile  = blockIdx.y;
  int b      = blockIdx.z;
  const __bf16* Bp = Xbf + (size_t)b * NN * Kd;
  fx8 acc[NP] = {};
  wmma_mm4(A, Kd, Bp, Kd, Kd, mtile, ntile0 * 16, lane, acc);
  int kh    = lane >> 4;
  int dbase = mtile * 16 + 8 * kh;
#pragma unroll
  for (int t = 0; t < NP; ++t) {
    int n = (ntile0 + t) * 16 + (lane & 15);
    if (mode == 0) {
#pragma unroll
      for (int r = 0; r < 8; ++r)
        outF[((size_t)b * M + dbase + r) * NN + n] = acc[t][r];
    } else if (mode == 1) {
      bfx8 o;
#pragma unroll
      for (int r = 0; r < 8; ++r) {
        int d = dbase + r;
        float s = bg[d] * rsqrtf(bvv[d] + EPS);
        float y = (acc[t][r] - bm[d]) * s + bbp[d];
        o[r] = (__bf16)fmaxf(y, 0.0f);
      }
      *(bfx8*)(outBF + ((size_t)b * NN + n) * M + dbase) = o;
    } else {
      bfx8 o;
#pragma unroll
      for (int r = 0; r < 8; ++r) {
        int d = dbase + r;
        float s  = bg[d] * rsqrtf(bvv[d] + EPS);
        float sh = bbp[d] - bm[d] * s;
        float xo = resid[((size_t)b * M + d) * NN + n];
        float y  = (xo + acc[t][r]) * s + sh;
        outF[((size_t)b * M + d) * NN + n] = y;
        o[r] = (__bf16)y;
      }
      *(bfx8*)(outBF + ((size_t)b * NN + n) * M + dbase) = o;
    }
  }
}

// ---------------------------------------------------------------------------
// Fused attention: 4 points per block. Block = 192 threads (6 waves = 6 M-tiles
// of D=96); each wave computes its M-tile for all 4 points as a 64-column B
// panel in LDS, reusing each A (weight) fragment across 4 WMMAs.
// Per point: pe1 -> bn/relu -> [WMMA pe_w2] -> a0 -> [WMMA am_w1] -> bn/relu ->
// [WMMA am_w2] -> softmax(K) -> weighted sum -> n1 BN residual -> x.
// ---------------------------------------------------------------------------
__global__ void attn_kernel(const float* __restrict__ points,
                            const int* __restrict__ idxw,
                            const float* __restrict__ pe_w1,
                            const float* __restrict__ pe_g, const float* __restrict__ pe_b,
                            const float* __restrict__ pe_m, const float* __restrict__ pe_v,
                            const __bf16* __restrict__ pew2,
                            const __bf16* __restrict__ amw1,
                            const __bf16* __restrict__ amw2,
                            const float* __restrict__ am_g, const float* __restrict__ am_b,
                            const float* __restrict__ am_m, const float* __restrict__ am_v,
                            const float* __restrict__ n1_g, const float* __restrict__ n1_b,
                            const float* __restrict__ n1_m, const float* __restrict__ n1_v,
                            const float* __restrict__ qf, const float* __restrict__ kf,
                            const float* __restrict__ vf,
                            float* __restrict__ xf, __bf16* __restrict__ xbf) {
  __shared__ int   s_idx[NP][KK];
  __shared__ float s_pd[NP][3][KK];
  __shared__ __attribute__((aligned(32))) __bf16 s_act[NP * KK * DD]; // [(p*16+col)*96+d]
  int n0 = blockIdx.x * NP, b = blockIdx.y;
  int tid = threadIdx.x;

  if (tid < NP * KK) {
    int p = tid >> 4, kc = tid & 15;
    s_idx[p][kc] = idxw[((size_t)b * NN + n0 + p) * KK + kc];
  }
  __syncthreads();
  {
    // 4 points * 48 = 192 pos_diff elements: exactly one per thread
    int p = tid / 48, r = tid % 48;
    int c = r % 3, col = r / 3;
    s_pd[p][c][col] = points[((size_t)b * NN + n0 + p) * 3 + c] -
                      points[((size_t)b * NN + s_idx[p][col]) * 3 + c];
  }
  __syncthreads();

  // t1 = relu(bn(pe_w1 @ pos_diff)) -> s_act (bf16)
  for (int e = tid; e < NP * DD * KK; e += 192) {
    int p = e / (DD * KK), rr = e % (DD * KK);
    int d = rr % DD, col = rr / DD;
    float pv = pe_w1[d * 3 + 0] * s_pd[p][0][col] +
               pe_w1[d * 3 + 1] * s_pd[p][1][col] +
               pe_w1[d * 3 + 2] * s_pd[p][2][col];
    float s = pe_g[d] * rsqrtf(pe_v[d] + EPS);
    float y = (pv - pe_m[d]) * s + pe_b[d];
    s_act[(p * KK + col) * DD + d] = (__bf16)fmaxf(y, 0.0f);
  }
  __syncthreads();

  int lane = tid & 31, m = tid >> 5;
  int kh = lane >> 4, col = lane & 15;
  int dbase = m * 16 + 8 * kh;

  // pe = pe_w2 @ t1 for all 4 points (kept in registers)
  fx8 accPE[NP] = {};
  wmma_mm4(pew2, DD, s_act, DD, DD, m, 0, lane, accPE);
  __syncthreads();

  // a0 = q - k_nn + pe -> s_act
#pragma unroll
  for (int p = 0; p < NP; ++p) {
    int nb = s_idx[p][col];
    bfx8 o;
#pragma unroll
    for (int r = 0; r < 8; ++r) {
      int d = dbase + r;
      float a0 = qf[((size_t)b * DD + d) * NN + (n0 + p)] -
                 kf[((size_t)b * DD + d) * NN + nb] + accPE[p][r];
      o[r] = (__bf16)a0;
    }
    *(bfx8*)(s_act + (p * KK + col) * DD + dbase) = o;
  }
  __syncthreads();

  fx8 acc1[NP] = {};
  wmma_mm4(amw1, DD, s_act, DD, DD, m, 0, lane, acc1);
  __syncthreads();

  // relu(bn(a1)) -> s_act
#pragma unroll
  for (int p = 0; p < NP; ++p) {
    bfx8 o;
#pragma unroll
    for (int r = 0; r < 8; ++r) {
      int d = dbase + r;
      float s = am_g[d] * rsqrtf(am_v[d] + EPS);
      float y = (acc1[p][r] - am_m[d]) * s + am_b[d];
      o[r] = (__bf16)fmaxf(y, 0.0f);
    }
    *(bfx8*)(s_act + (p * KK + col) * DD + dbase) = o;
  }
  __syncthreads();

  fx8 acc2[NP] = {};
  wmma_mm4(amw2, DD, s_act, DD, DD, m, 0, lane, acc2);

  // softmax over K (16 columns live in the 16 lanes of each half-wave group),
  // weighted sum, n1 BN + residual.
#pragma unroll
  for (int p = 0; p < NP; ++p) {
    int nb = s_idx[p][col];
#pragma unroll
    for (int r = 0; r < 8; ++r) {
      int d = dbase + r;
      float a = acc2[p][r];
      float mx = a;
      for (int off = 1; off < 16; off <<= 1) mx = fmaxf(mx, __shfl_xor(mx, off, 32));
      float e = __expf(a - mx);
      float ssum = e;
      for (int off = 1; off < 16; off <<= 1) ssum += __shfl_xor(ssum, off, 32);
      float attn = e / ssum;
      float vv = vf[((size_t)b * DD + d) * NN + nb];
      float w = attn * (vv + accPE[p][r]);
      for (int off = 1; off < 16; off <<= 1) w += __shfl_xor(w, off, 32);
      if (col == 0) {
        float s  = n1_g[d] * rsqrtf(n1_v[d] + EPS);
        float sh = n1_b[d] - n1_m[d] * s;
        size_t xi = ((size_t)b * DD + d) * NN + (n0 + p);
        float y = (xf[xi] + w) * s + sh;
        xf[xi] = y;
        xbf[((size_t)b * NN + (n0 + p)) * DD + d] = (__bf16)y;
      }
    }
  }
}

// ---------------------------------------------------------------------------
// Head tail: out[b,o,n] = hd_w2[o,:] . h[b,n,:] + hd_b2[o]   (o = 0..1)
// ---------------------------------------------------------------------------
__global__ void head2_kernel(const __bf16* __restrict__ hbf,
                             const float* __restrict__ w2,
                             const float* __restrict__ b2,
                             float* __restrict__ out) {
  int n = blockIdx.x * 256 + threadIdx.x;
  int b = blockIdx.y;
  const __bf16* h = hbf + ((size_t)b * NN + n) * 128;
  float a0 = b2[0], a1 = b2[1];
  for (int c = 0; c < 128; ++c) {
    float hv = (float)h[c];
    a0 += w2[c] * hv;
    a1 += w2[128 + c] * hv;
  }
  out[((size_t)b * NCC + 0) * NN + n] = a0;
  out[((size_t)b * NCC + 1) * NN + n] = a1;
}

// ---------------------------------------------------------------------------
// Host orchestration
// ---------------------------------------------------------------------------
extern "C" void kernel_launch(void* const* d_in, const int* in_sizes, int n_in,
                              void* d_out, int out_size, void* d_ws, size_t ws_size,
                              hipStream_t stream) {
  const float* points  = (const float*)d_in[0];
  const float* embed_w = (const float*)d_in[1];
  const float* embed_g = (const float*)d_in[2];
  const float* embed_b = (const float*)d_in[3];
  const float* embed_m = (const float*)d_in[4];
  const float* embed_v = (const float*)d_in[5];
  const float* wq   = (const float*)d_in[6];
  const float* wk   = (const float*)d_in[7];
  const float* wv   = (const float*)d_in[8];
  const float* pe_w1 = (const float*)d_in[9];
  const float* pe_g  = (const float*)d_in[10];
  const float* pe_b  = (const float*)d_in[11];
  const float* pe_m  = (const float*)d_in[12];
  const float* pe_v  = (const float*)d_in[13];
  const float* pe_w2 = (const float*)d_in[14];
  const float* am_w1 = (const float*)d_in[15];
  const float* am_g  = (const float*)d_in[16];
  const float* am_b  = (const float*)d_in[17];
  const float* am_m  = (const float*)d_in[18];
  const float* am_v  = (const float*)d_in[19];
  const float* am_w2 = (const float*)d_in[20];
  const float* n1_g  = (const float*)d_in[21];
  const float* n1_b  = (const float*)d_in[22];
  const float* n1_m  = (const float*)d_in[23];
  const float* n1_v  = (const float*)d_in[24];
  const float* ffn_w1 = (const float*)d_in[25];
  const float* f_g   = (const float*)d_in[26];
  const float* f_b   = (const float*)d_in[27];
  const float* f_m   = (const float*)d_in[28];
  const float* f_v   = (const float*)d_in[29];
  const float* ffn_w2 = (const float*)d_in[30];
  const float* n2_g  = (const float*)d_in[31];
  const float* n2_b  = (const float*)d_in[32];
  const float* n2_m  = (const float*)d_in[33];
  const float* n2_v  = (const float*)d_in[34];
  const float* hd_w1 = (const float*)d_in[35];
  const float* hd_g  = (const float*)d_in[36];
  const float* hd_b  = (const float*)d_in[37];
  const float* hd_m  = (const float*)d_in[38];
  const float* hd_v  = (const float*)d_in[39];
  const float* hd_w2 = (const float*)d_in[40];
  const float* hd_b2 = (const float*)d_in[41];

  // ------- workspace carve-up -------
  char* wsp = (char*)d_ws;
  auto alloc = [&](size_t bytes) {
    void* r = (void*)wsp;
    wsp += (bytes + 255) & ~(size_t)255;
    return r;
  };
  float*  x_f   = (float*) alloc((size_t)BB * DD * NN * 4);
  __bf16* x_bf  = (__bf16*)alloc((size_t)BB * NN * DD * 2);
  float*  q_f   = (float*) alloc((size_t)BB * DD * NN * 4);
  float*  k_f   = (float*) alloc((size_t)BB * DD * NN * 4);
  float*  v_f   = (float*) alloc((size_t)BB * DD * NN * 4);
  __bf16* h_bf  = (__bf16*)alloc((size_t)BB * NN * HH * 2);
  __bf16* hd_bf = (__bf16*)alloc((size_t)BB * NN * 128 * 2);
  int*    idxw  = (int*)   alloc((size_t)BB * NN * KK * 4);
  __bf16* wq_bf  = (__bf16*)alloc((size_t)LL * DD * DD * 2);
  __bf16* wk_bf  = (__bf16*)alloc((size_t)LL * DD * DD * 2);
  __bf16* wv_bf  = (__bf16*)alloc((size_t)LL * DD * DD * 2);
  __bf16* pw2_bf = (__bf16*)alloc((size_t)LL * DD * DD * 2);
  __bf16* aw1_bf = (__bf16*)alloc((size_t)LL * DD * DD * 2);
  __bf16* aw2_bf = (__bf16*)alloc((size_t)LL * DD * DD * 2);
  __bf16* fw1_bf = (__bf16*)alloc((size_t)LL * HH * DD * 2);
  __bf16* fw2_bf = (__bf16*)alloc((size_t)LL * DD * HH * 2);
  __bf16* hw1_bf = (__bf16*)alloc((size_t)128 * DD * 2);

  // ------- weight conversion to bf16 -------
  auto cvt = [&](const float* s, __bf16* d, int n) {
    cvt_bf16_kernel<<<(n + 255) / 256, 256, 0, stream>>>(s, d, n);
  };
  cvt(wq,     wq_bf,  LL * DD * DD);
  cvt(wk,     wk_bf,  LL * DD * DD);
  cvt(wv,     wv_bf,  LL * DD * DD);
  cvt(pe_w2,  pw2_bf, LL * DD * DD);
  cvt(am_w1,  aw1_bf, LL * DD * DD);
  cvt(am_w2,  aw2_bf, LL * DD * DD);
  cvt(ffn_w1, fw1_bf, LL * HH * DD);
  cvt(ffn_w2, fw2_bf, LL * DD * HH);
  cvt(hd_w1,  hw1_bf, 128 * DD);

  // ------- embed + knn -------
  embed_kernel<<<dim3(NN / 256, DD, BB), 256, 0, stream>>>(
      points, embed_w, embed_g, embed_b, embed_m, embed_v, x_f, x_bf);
  knn_kernel<<<dim3(BB * NN / 256), 256, 0, stream>>>(points, idxw);

  // ------- transformer layers -------
  // gemm grid.x: N / (16 cols * NP tiles * 4 waves) = 8192/256 = 32
  for (int l = 0; l < LL; ++l) {
    const __bf16* wql = wq_bf + (size_t)l * DD * DD;
    const __bf16* wkl = wk_bf + (size_t)l * DD * DD;
    const __bf16* wvl = wv_bf + (size_t)l * DD * DD;
    gemm_kernel<<<dim3(NN / 256, DD / 16, BB), 128, 0, stream>>>(
        wql, x_bf, DD, DD, 0, nullptr, nullptr, nullptr, nullptr, nullptr, q_f, nullptr);
    gemm_kernel<<<dim3(NN / 256, DD / 16, BB), 128, 0, stream>>>(
        wkl, x_bf, DD, DD, 0, nullptr, nullptr, nullptr, nullptr, nullptr, k_f, nullptr);
    gemm_kernel<<<dim3(NN / 256, DD / 16, BB), 128, 0, stream>>>(
        wvl, x_bf, DD, DD, 0, nullptr, nullptr, nullptr, nullptr, nullptr, v_f, nullptr);

    attn_kernel<<<dim3(NN / NP, BB), 192, 0, stream>>>(
        points, idxw, pe_w1 + (size_t)l * DD * 3,
        pe_g + l * DD, pe_b + l * DD, pe_m + l * DD, pe_v + l * DD,
        pw2_bf + (size_t)l * DD * DD, aw1_bf + (size_t)l * DD * DD,
        aw2_bf + (size_t)l * DD * DD,
        am_g + l * DD, am_b + l * DD, am_m + l * DD, am_v + l * DD,
        n1_g + l * DD, n1_b + l * DD, n1_m + l * DD, n1_v + l * DD,
        q_f, k_f, v_f, x_f, x_bf);

    // FFN1: h = relu(bn(ffn_w1 @ x))  (M=192, Kd=96), mode 1 -> h_bf
    gemm_kernel<<<dim3(NN / 256, HH / 16, BB), 128, 0, stream>>>(
        fw1_bf + (size_t)l * HH * DD, x_bf, HH, DD, 1,
        f_g + l * HH, f_b + l * HH, f_m + l * HH, f_v + l * HH,
        nullptr, nullptr, h_bf);
    // FFN2 + residual + n2 BN: x = bn(x + ffn_w2 @ h) (M=96, Kd=192), mode 2
    gemm_kernel<<<dim3(NN / 256, DD / 16, BB), 128, 0, stream>>>(
        fw2_bf + (size_t)l * DD * HH, h_bf, DD, HH, 2,
        n2_g + l * DD, n2_b + l * DD, n2_m + l * DD, n2_v + l * DD,
        x_f, x_f, x_bf);
  }

  // ------- head -------
  gemm_kernel<<<dim3(NN / 256, 128 / 16, BB), 128, 0, stream>>>(
      hw1_bf, x_bf, 128, DD, 1, hd_g, hd_b, hd_m, hd_v,
      nullptr, nullptr, hd_bf);
  head2_kernel<<<dim3(NN / 256, BB), 256, 0, stream>>>(
      hd_bf, hd_w2, hd_b2, (float*)d_out);
}
